// Model_1443109011556
// MI455X (gfx1250) — matrix operand order repaired and checked by reading, not compiled, not run on hardware
//
#include <hip/hip_runtime.h>
#include <hip/hip_bf16.h>

// Fused 2-layer LSTM (B=131072, T=30, Din=88, H=20) + FC + softmax for gfx1250.
// Strategy: read x exactly once (1.38 GB -> ~59us HBM floor), keep all recurrent
// state in registers/LDS, do every matmul with v_wmma_f32_16x16x32_f16.
// H padded to 32 so gates form 8 aligned 16-wide N-tiles (i,f,g,o stay in the
// same lane/VGPR slot -> pure per-lane cell math, no cross-lane shuffles).

typedef __attribute__((ext_vector_type(16))) _Float16 v16h;
typedef __attribute__((ext_vector_type(8)))  _Float16 v8h;
typedef __attribute__((ext_vector_type(8)))  float    v8f;
typedef __attribute__((ext_vector_type(4)))  float    f32x4;

namespace {
constexpr int kB    = 131072;
constexpr int kT    = 30;
constexpr int kDin  = 88;
constexpr int kH    = 20;
constexpr int kHP   = 32;          // padded hidden
constexpr int kNT   = 8;           // N tiles per projection (128 padded gates / 16)
constexpr int kWPW  = 8;           // waves per workgroup
constexpr int kTilesIh0 = 3 * kNT; // 24 tiles: x-projection (K = 96 = 3 chunks)
constexpr int kNTiles   = kTilesIh0 + 3 * kNT; // + W_hh0, W_ih1, W_hh1 = 48
}

__device__ __forceinline__ float fast_exp(float x) {
  return __builtin_amdgcn_exp2f(x * 1.44269504088896f);
}
__device__ __forceinline__ float fast_sig(float x) {
#if __has_builtin(__builtin_amdgcn_tanhf)
  return fmaf(0.5f, __builtin_amdgcn_tanhf(0.5f * x), 0.5f);  // 1 trans op
#else
  return __builtin_amdgcn_rcpf(1.0f + fast_exp(-x));          // 2 trans ops
#endif
}
__device__ __forceinline__ float fast_tanh(float x) {
#if __has_builtin(__builtin_amdgcn_tanhf)
  return __builtin_amdgcn_tanhf(x);
#else
  return fmaf(2.0f, fast_sig(2.0f * x), -1.0f);
#endif
}

__device__ __forceinline__ v8f wmma16(v16h a, v16h b, v8f c) {
  return __builtin_amdgcn_wmma_f32_16x16x32_f16(false, a, false, b, (short)0, c,
                                                false, false);
}

__global__ __launch_bounds__(kWPW * 32) void lstm2_fused(
    const float* __restrict__ x,
    const float* __restrict__ W_ih0, const float* __restrict__ W_hh0,
    const float* __restrict__ b_ih0, const float* __restrict__ b_hh0,
    const float* __restrict__ W_ih1, const float* __restrict__ W_hh1,
    const float* __restrict__ b_ih1, const float* __restrict__ b_hh1,
    const float* __restrict__ W_fc,  const float* __restrict__ b_fc,
    float* __restrict__ out)
{
  // 48 KB of prepacked f16 weight tiles in WMMA B layout:
  //   lane L -> column N = L%16 ; element e -> K = (e%8) + 8*(L/16) + 16*(e/8)
  __shared__ __align__(32) _Float16 wt[kNTiles][32][16];
  __shared__ float bls[2][128];                       // combined biases (padded)
  __shared__ __align__(32) _Float16 hst[kWPW][16][kHP]; // per-wave h transpose

  const int tid = threadIdx.x;

  // ---------------- cooperative weight packing (once per WG) ----------------
  for (int idx = tid; idx < kNTiles * 512; idx += blockDim.x) {
    const int tile = idx >> 9;
    const int rem  = idx & 511;
    const int ln   = rem >> 4;         // lane this element belongs to
    const int e    = rem & 15;         // v16h element
    const int N    = ln & 15;
    const int kl   = (e & 7) + ((ln >> 4) << 3) + ((e >> 3) << 4);  // K within chunk
    int proj, kc, nt;
    if (tile < kTilesIh0)                { proj = 0; kc = tile >> 3; nt = tile & 7; }
    else if (tile < kTilesIh0 + 8)       { proj = 1; kc = 0; nt = tile - kTilesIh0; }
    else if (tile < kTilesIh0 + 16)      { proj = 2; kc = 0; nt = tile - kTilesIh0 - 8; }
    else                                 { proj = 3; kc = 0; nt = tile - kTilesIh0 - 16; }
    const int G    = nt * 16 + N;      // padded gate index 0..127
    const int grp  = G >> 5;           // 0=i 1=f 2=g 3=o
    const int unit = G & 31;
    const int K    = kc * 32 + kl;
    float v = 0.0f;
    if (unit < kH) {
      const int rg = grp * kH + unit;  // real PyTorch gate row
      if      (proj == 0) { if (K < kDin) v = W_ih0[rg * kDin + K]; }
      else if (proj == 1) { if (K < kH)   v = W_hh0[rg * kH + K]; }
      else if (proj == 2) { if (K < kH)   v = W_ih1[rg * kH + K]; }
      else                { if (K < kH)   v = W_hh1[rg * kH + K]; }
    }
    wt[tile][ln][e] = (_Float16)v;
  }
  for (int g = tid; g < 256; g += blockDim.x) {
    const int layer = g >> 7;
    const int G = g & 127;
    const int grp = G >> 5, unit = G & 31;
    float v = 0.0f;
    if (unit < kH) {
      const int rg = grp * kH + unit;
      v = layer ? (b_ih1[rg] + b_hh1[rg]) : (b_ih0[rg] + b_hh0[rg]);
    }
    bls[layer][G] = v;
  }
  __syncthreads();

  // ---------------- per-wave setup ----------------
  const int wave = tid >> 5;
  const int lane = tid & 31;
  const int lo   = lane & 15;   // sample-in-tile (A rows / output N)
  const int hi   = lane >> 4;   // lane half
  const long m   = (long)(blockIdx.x * kWPW + wave) * 16 + lo;

  float bias0[8], bias1[8];     // [grp*2 + tile_half], uniform across VGPR r
#pragma unroll
  for (int grp = 0; grp < 4; ++grp)
#pragma unroll
    for (int p = 0; p < 2; ++p) {
      bias0[grp * 2 + p] = bls[0][grp * 32 + p * 16 + lo];
      bias1[grp * 2 + p] = bls[1][grp * 32 + p * 16 + lo];
    }

  v16h hAa = {};                // layer0 h, WMMA A layout (f16)
  v16h hBa = {};                // layer1 h, WMMA A layout (f16)
  v8f  cA[2] = {};              // layer0 c, D layout (units 0..15 / 16..31)
  v8f  cB[2] = {};
  _Float16* const myst = &hst[wave][0][0];
  const float* xrow = x + m * (long)(kT * kDin);

  for (int t = 0; t < kT; ++t, xrow += kDin) {
    // ---- x_t into 3 A-operand chunks (f32 global -> f16 regs) ----
    // A layout: lane L row M=L%16, element e -> K = c*32 + e + 8*(e/8) + 8*(L/16)
    v16h xa[3];
#pragma unroll
    for (int c = 0; c < 3; ++c) {
      const int k0 = c * 32 + 8 * hi;
      const int k1 = k0 + 16;
      f32x4 a0 = *(const f32x4*)(xrow + k0);
      f32x4 a1 = *(const f32x4*)(xrow + k0 + 4);
      const bool ok = (k1 < kDin);          // only false: c==2, hi==1 (K=88..95)
      const int k1c = ok ? k1 : 0;          // branchless, stays in bounds
      f32x4 b0 = *(const f32x4*)(xrow + k1c);
      f32x4 b1 = *(const f32x4*)(xrow + k1c + 4);
      const float s = ok ? 1.0f : 0.0f;
      v16h r;
      r[0]=(_Float16)a0.x;  r[1]=(_Float16)a0.y;  r[2]=(_Float16)a0.z;  r[3]=(_Float16)a0.w;
      r[4]=(_Float16)a1.x;  r[5]=(_Float16)a1.y;  r[6]=(_Float16)a1.z;  r[7]=(_Float16)a1.w;
      r[8]=(_Float16)(b0.x*s);  r[9]=(_Float16)(b0.y*s);  r[10]=(_Float16)(b0.z*s); r[11]=(_Float16)(b0.w*s);
      r[12]=(_Float16)(b1.x*s); r[13]=(_Float16)(b1.y*s); r[14]=(_Float16)(b1.z*s); r[15]=(_Float16)(b1.w*s);
      xa[c] = r;
    }

    // ---- layer 0: gates = x*W_ih0^T + hA*W_hh0^T  (32 WMMAs) ----
    v8f acc[8];
#pragma unroll
    for (int n = 0; n < kNT; ++n) {
      v8f cc = {};
      cc = wmma16(xa[0], *(const v16h*)&wt[0 * 8 + n][lane][0], cc);
      cc = wmma16(xa[1], *(const v16h*)&wt[1 * 8 + n][lane][0], cc);
      cc = wmma16(xa[2], *(const v16h*)&wt[2 * 8 + n][lane][0], cc);
      cc = wmma16(hAa,   *(const v16h*)&wt[24 + n][lane][0],    cc);
      acc[n] = cc;
    }
    // cell math (per-lane; i/f/g/o aligned across tiles) + stage hA as f16
#pragma unroll
    for (int p = 0; p < 2; ++p) {
#pragma unroll
      for (int r = 0; r < 8; ++r) {
        const float iv = fast_sig (acc[0 + p][r] + bias0[0 + p]);
        const float fv = fast_sig (acc[2 + p][r] + bias0[2 + p]);
        const float gv = fast_tanh(acc[4 + p][r] + bias0[4 + p]);
        const float ov = fast_sig (acc[6 + p][r] + bias0[6 + p]);
        const float cn = fmaf(fv, cA[p][r], iv * gv);
        cA[p][r] = cn;
        const float hv = ov * fast_tanh(cn);
        myst[(r + 8 * hi) * kHP + (p * 16 + lo)] = (_Float16)hv; // [sample][unit]
      }
    }
    asm volatile("" ::: "memory");  // LDS is in-order per wave; block reordering
    {   // transpose-read hA into WMMA A layout
      const _Float16* rp = myst + lo * kHP + 8 * hi;
      v8h l0 = *(const v8h*)(rp);
      v8h l1 = *(const v8h*)(rp + 16);
#pragma unroll
      for (int e = 0; e < 8; ++e) { hAa[e] = l0[e]; hAa[8 + e] = l1[e]; }
    }

    // ---- layer 1: gates = hA*W_ih1^T + hB*W_hh1^T  (16 WMMAs) ----
#pragma unroll
    for (int n = 0; n < kNT; ++n) {
      v8f cc = {};
      cc = wmma16(hAa, *(const v16h*)&wt[32 + n][lane][0], cc);
      cc = wmma16(hBa, *(const v16h*)&wt[40 + n][lane][0], cc);
      acc[n] = cc;
    }
    asm volatile("" ::: "memory");
#pragma unroll
    for (int p = 0; p < 2; ++p) {
#pragma unroll
      for (int r = 0; r < 8; ++r) {
        const float iv = fast_sig (acc[0 + p][r] + bias1[0 + p]);
        const float fv = fast_sig (acc[2 + p][r] + bias1[2 + p]);
        const float gv = fast_tanh(acc[4 + p][r] + bias1[4 + p]);
        const float ov = fast_sig (acc[6 + p][r] + bias1[6 + p]);
        const float cn = fmaf(fv, cB[p][r], iv * gv);
        cB[p][r] = cn;
        const float hv = ov * fast_tanh(cn);
        myst[(r + 8 * hi) * kHP + (p * 16 + lo)] = (_Float16)hv;
      }
    }
    asm volatile("" ::: "memory");
    {
      const _Float16* rp = myst + lo * kHP + 8 * hi;
      v8h l0 = *(const v8h*)(rp);
      v8h l1 = *(const v8h*)(rp + 16);
#pragma unroll
      for (int e = 0; e < 8; ++e) { hBa[e] = l0[e]; hBa[8 + e] = l1[e]; }
    }
  }

  // ---------------- FC + 2-class softmax (myst holds final hB) --------------
  asm volatile("" ::: "memory");
  if (lane < 16) {
    const _Float16* hrow = myst + lo * kHP;
    float z0 = b_fc[0];
    float z1 = b_fc[1];
#pragma unroll
    for (int j = 0; j < kH; ++j) {
      const float hv = (float)hrow[j];
      z0 = fmaf(hv, W_fc[j],      z0);
      z1 = fmaf(hv, W_fc[kH + j], z1);
    }
    const float mx = fmaxf(z0, z1);
    const float e0 = fast_exp(z0 - mx);
    const float e1 = fast_exp(z1 - mx);
    const float inv = __builtin_amdgcn_rcpf(e0 + e1);
    out[m * 2 + 0] = e0 * inv;
    out[m * 2 + 1] = e1 * inv;
  }
}

extern "C" void kernel_launch(void* const* d_in, const int* in_sizes, int n_in,
                              void* d_out, int out_size, void* d_ws, size_t ws_size,
                              hipStream_t stream) {
  (void)in_sizes; (void)n_in; (void)out_size; (void)d_ws; (void)ws_size;
  const float* x     = (const float*)d_in[0];
  const float* W_ih0 = (const float*)d_in[1];
  const float* W_hh0 = (const float*)d_in[2];
  const float* b_ih0 = (const float*)d_in[3];
  const float* b_hh0 = (const float*)d_in[4];
  const float* W_ih1 = (const float*)d_in[5];
  const float* W_hh1 = (const float*)d_in[6];
  const float* b_ih1 = (const float*)d_in[7];
  const float* b_hh1 = (const float*)d_in[8];
  const float* W_fc  = (const float*)d_in[9];
  const float* b_fc  = (const float*)d_in[10];
  float* out = (float*)d_out;

  const int grid = kB / (16 * kWPW);  // 1024 WGs x 256 threads = 8192 waves x 16 rows
  lstm2_fused<<<dim3(grid), dim3(kWPW * 32), 0, stream>>>(
      x, W_ih0, W_hh0, b_ih0, b_hh0, W_ih1, W_hh1, b_ih1, b_hh1, W_fc, b_fc, out);
}
